// WrapperLinear_10024453669435
// MI455X (gfx1250) — matrix operand order repaired and checked by reading, not compiled, not run on hardware
//
#include <hip/hip_runtime.h>
#include <stdint.h>

typedef __attribute__((ext_vector_type(4)))  _Float16 v4h;
typedef __attribute__((ext_vector_type(8)))  _Float16 v8h;
typedef __attribute__((ext_vector_type(16))) _Float16 v16h;
typedef __attribute__((ext_vector_type(8)))  float    v8f;

#define O_DIM 4096
#define I_DIM 4096
#define M_DIM 4096   /* B*S = 2*2048 */
#define GROUP 128
#define MAXQF 15.0f

#define BM 128
#define BN 128
#define BK 32
#define PITCH 40                 /* halfwords per LDS row (BK + 8 pad), 80B: conflict-free */
#define NKT (I_DIM / BK)         /* 128 K-steps */

// ---------------------------------------------------------------------------
// CDNA5 async global->LDS copy (ASYNCcnt-tracked), per CDNA5 ISA.
// VDST = LDS byte offset (VGPR), VADDR = 64-bit global address, GV mode.
// ---------------------------------------------------------------------------
__device__ __forceinline__ void async_b128(uint32_t lds_off, const void* gptr) {
    asm volatile("global_load_async_to_lds_b128 %0, %1, off"
                 :: "v"(lds_off), "v"((unsigned long long)(uintptr_t)gptr)
                 : "memory");
}

__device__ __forceinline__ void wait_async0() {
#if __has_builtin(__builtin_amdgcn_s_wait_asynccnt)
    __builtin_amdgcn_s_wait_asynccnt(0);
#else
    asm volatile("s_wait_asynccnt 0x0" ::: "memory");
#endif
}

__device__ __forceinline__ v16h ld_frag(const _Float16* p0, const _Float16* p1) {
    v8h lo = *(const v8h*)p0;   // ds_load_b128
    v8h hi = *(const v8h*)p1;   // ds_load_b128
    return __builtin_shufflevector(lo, hi, 0,1,2,3,4,5,6,7,8,9,10,11,12,13,14,15);
}

// ---------------------------------------------------------------------------
// Kernel 1: group-wise asymmetric 4-bit fake-quant of weight -> f16.
// One wave32 per group of 128 contiguous elements (4 per lane).
// ---------------------------------------------------------------------------
__global__ __launch_bounds__(256) void quant_kernel(
        const float* __restrict__ w, const float* __restrict__ v,
        const float* __restrict__ min_scale, const float* __restrict__ max_scale,
        _Float16* __restrict__ wq)
{
    const int warp = threadIdx.x >> 5;
    const int lane = threadIdx.x & 31;
    const int g = blockIdx.x * 8 + warp;
    const size_t base = (size_t)g * GROUP + (size_t)lane * 4;

    float4 w4 = *(const float4*)(w + base);
    float4 v4 = *(const float4*)(v + base);

    float mn = fminf(fminf(w4.x, w4.y), fminf(w4.z, w4.w));
    float mx = fmaxf(fmaxf(w4.x, w4.y), fmaxf(w4.z, w4.w));
#pragma unroll
    for (int off = 16; off; off >>= 1) {
        mn = fminf(mn, __shfl_xor(mn, off, 32));
        mx = fmaxf(mx, __shfl_xor(mx, off, 32));
    }

    float wmin_t = fminf(mn, 0.0f);
    float wmax_t = fmaxf(mx, 0.0f);
    float ms = fminf(fmaxf(min_scale[g], -1.0f), 0.0f) + 1.0f;
    float xs = fminf(fmaxf(max_scale[g], -1.0f), 0.0f) + 1.0f;
    float wmin = wmin_t * ms;
    float wmax = wmax_t * xs;
    if (wmin == 0.0f && wmax == 0.0f) { wmin = -1.0f; wmax = 1.0f; }
    float scale = (float)(_Float16)((wmax - wmin) / MAXQF);  // fp16-precision scale
    float zp = rintf(-wmin / scale);                          // round-to-nearest-even

    float we[4] = {w4.x, w4.y, w4.z, w4.w};
    float ve[4] = {v4.x, v4.y, v4.z, v4.w};
    v4h outv;
#pragma unroll
    for (int r = 0; r < 4; ++r) {
        float iw = rintf(we[r] / scale + ve[r]);
        float q  = fminf(fmaxf(iw + zp, 0.0f), MAXQF);
        outv[r] = (_Float16)(scale * (q - zp));
    }
    *(v4h*)(wq + base) = outv;
}

// ---------------------------------------------------------------------------
// Kernel 2: x fp32 -> f16 (vectorized elementwise).
// ---------------------------------------------------------------------------
__global__ __launch_bounds__(256) void cvt_kernel(
        const float* __restrict__ x, _Float16* __restrict__ xf)
{
    const size_t i = ((size_t)blockIdx.x * blockDim.x + threadIdx.x) * 4;
    float4 f = *(const float4*)(x + i);
    v4h h = {(_Float16)f.x, (_Float16)f.y, (_Float16)f.z, (_Float16)f.w};
    *(v4h*)(xf + i) = h;
}

// ---------------------------------------------------------------------------
// Kernel 3: C[M,N] = A[M,K] * B[N,K]^T + bias, f16 WMMA, f32 accumulate.
// 128x128x32 tiles, async global->LDS double buffering, 8 waves/block.
// ---------------------------------------------------------------------------
__global__ __launch_bounds__(256) void gemm_kernel(
        const _Float16* __restrict__ A,    // [M_DIM, I_DIM]
        const _Float16* __restrict__ Bw,   // [O_DIM, I_DIM]
        const float* __restrict__ bias,    // [O_DIM]
        float* __restrict__ Cout)          // [M_DIM, O_DIM]
{
    __shared__ __align__(128) _Float16 smem[2][(BM + BN) * PITCH];

    const int tid = threadIdx.x;
    const int bx = blockIdx.x;   // N tile
    const int by = blockIdx.y;   // M tile

    // --- staging: 512 16B chunks per 128x32 tile; 2 A-chunks + 2 B-chunks/thread
    const int c0 = tid, c1 = tid + 256;
    const int r0 = c0 >> 2, col0 = (c0 & 3) << 3;
    const int r1 = c1 >> 2, col1 = (c1 & 3) << 3;

    const _Float16* gA0 = A  + ((size_t)(by * BM + r0)) * I_DIM + col0;
    const _Float16* gA1 = A  + ((size_t)(by * BM + r1)) * I_DIM + col1;
    const _Float16* gB0 = Bw + ((size_t)(bx * BN + r0)) * I_DIM + col0;
    const _Float16* gB1 = Bw + ((size_t)(bx * BN + r1)) * I_DIM + col1;

    uint32_t sA0[2], sA1[2], sB0[2], sB1[2];
#pragma unroll
    for (int b = 0; b < 2; ++b) {
        sA0[b] = (uint32_t)(uintptr_t)&smem[b][r0 * PITCH + col0];
        sA1[b] = (uint32_t)(uintptr_t)&smem[b][r1 * PITCH + col1];
        sB0[b] = (uint32_t)(uintptr_t)&smem[b][BM * PITCH + r0 * PITCH + col0];
        sB1[b] = (uint32_t)(uintptr_t)&smem[b][BM * PITCH + r1 * PITCH + col1];
    }

    auto issue = [&](int buf, int kt) {
        const size_t ko = (size_t)kt * BK;
        async_b128(sA0[buf], gA0 + ko);
        async_b128(sA1[buf], gA1 + ko);
        async_b128(sB0[buf], gB0 + ko);
        async_b128(sB1[buf], gB1 + ko);
    };

    // --- wave / lane layout: 8 waves -> 2 (M) x 4 (N); each wave: 64x32 of C
    const int lane = tid & 31;
    const int wv = tid >> 5;
    const int wm = wv & 1;        // M half of block tile
    const int wn = wv >> 1;       // N quarter of block tile
    const int half = lane >> 4;
    const int m16 = lane & 15;

    v8f acc[4][2] = {};

    issue(0, 0);
    for (int kt = 0; kt < NKT; ++kt) {
        wait_async0();            // our async writes for buf[kt&1] done
        __syncthreads();          // everyone's writes visible; prev reads done
        if (kt + 1 < NKT) issue((kt + 1) & 1, kt + 1);  // overlap with compute

        const _Float16* As = &smem[kt & 1][0];
        const _Float16* Bs = As + BM * PITCH;

        v16h a[4], b[2];
#pragma unroll
        for (int i = 0; i < 4; ++i) {
            const _Float16* p = As + (wm * 64 + i * 16 + m16) * PITCH + half * 8;
            a[i] = ld_frag(p, p + 16);   // K {0..7,16..23} / {8..15,24..31}
        }
#pragma unroll
        for (int j = 0; j < 2; ++j) {
            const _Float16* p = Bs + (wn * 32 + j * 16 + m16) * PITCH + half * 8;
            b[j] = ld_frag(p, p + 16);   // K {0..7,16..23} / {8..15,24..31}, N = m16
        }
#pragma unroll
        for (int i = 0; i < 4; ++i)
#pragma unroll
            for (int j = 0; j < 2; ++j)
                acc[i][j] = __builtin_amdgcn_wmma_f32_16x16x32_f16(
                    false, a[i], false, b[j], (short)0, acc[i][j], false, false);
    }

    // --- epilogue: C layout lane(0-15)->M=r, lane(16-31)->M=r+8, N=m16
    const int nbase = bx * BN + wn * 32;
    float bj[2];
#pragma unroll
    for (int j = 0; j < 2; ++j) bj[j] = bias[nbase + j * 16 + m16];

#pragma unroll
    for (int i = 0; i < 4; ++i) {
#pragma unroll
        for (int j = 0; j < 2; ++j) {
            const int col = nbase + j * 16 + m16;
#pragma unroll
            for (int r = 0; r < 8; ++r) {
                const size_t row = (size_t)by * BM + wm * 64 + i * 16 + half * 8 + r;
                Cout[row * O_DIM + col] = acc[i][j][r] + bj[j];
            }
        }
    }
}

// ---------------------------------------------------------------------------
extern "C" void kernel_launch(void* const* d_in, const int* in_sizes, int n_in,
                              void* d_out, int out_size, void* d_ws, size_t ws_size,
                              hipStream_t stream) {
    const float* x      = (const float*)d_in[0];
    const float* weight = (const float*)d_in[1];
    const float* bias   = (const float*)d_in[2];
    const float* value  = (const float*)d_in[3];
    const float* minsc  = (const float*)d_in[4];
    const float* maxsc  = (const float*)d_in[5];
    float* out = (float*)d_out;

    _Float16* wq = (_Float16*)d_ws;                                       // 32 MB
    _Float16* xf = (_Float16*)((char*)d_ws + (size_t)O_DIM * I_DIM * 2);  // 32 MB

    quant_kernel<<<(O_DIM * I_DIM / GROUP) / 8, 256, 0, stream>>>(
        weight, value, minsc, maxsc, wq);
    cvt_kernel<<<((size_t)M_DIM * I_DIM / 4) / 256, 256, 0, stream>>>(x, xf);
    gemm_kernel<<<dim3(O_DIM / BN, M_DIM / BM), 256, 0, stream>>>(
        xf, wq, bias, out);
}
